// SpiralAutoencoder_2808908611875
// MI455X (gfx1250) — matrix operand order repaired and checked by reading, not compiled, not run on hardware
//
#include <hip/hip_runtime.h>
#include <math.h>

typedef __attribute__((ext_vector_type(16))) _Float16 v16h;
typedef __attribute__((ext_vector_type(8)))  float    v8f;

#define TM 64      // M tile (4 wave-rows of 16)
#define TN 32      // N tile (2 wave-cols of 16)
#define KC 32      // K chunk = one v_wmma_f32_16x16x32_f16
#define LDSW 34    // half-stride per LDS row (68B, bank stagger)
#define SP 9       // spiral length

union FragH { v16h v; unsigned u[8]; };

__device__ __forceinline__ unsigned pkh(float a, float b) {
    union { _Float16 h[2]; unsigned u; } t;
    t.h[0] = (_Float16)a; t.h[1] = (_Float16)b;
    return t.u;
}

// Fast layer-specialized GEMM. Requires M%64==0, K%32==0, (GATHER -> CIN%32==0).
// GATHER: A[m,k] = x[b, idx[vtx, k/CIN], k%CIN], b = m>>LGV, vtx = m&((1<<LGV)-1)
// dense : A[m,k] = x[m*K + k]
// wt pre-transposed: wt[n*K + k]
template<int K, int CIN, int LGV, bool GATHER, bool ELU, bool RES, bool NG>
__global__ __launch_bounds__(256)
void wmma_conv(const float* __restrict__ x,
               const int*   __restrict__ idx,
               const float* __restrict__ wt,
               const float* __restrict__ bias,
               const float* __restrict__ res,
               float* __restrict__ out,
               int Nout)
{
    __shared__ _Float16 As[2][TM][LDSW];
    __shared__ _Float16 Bs[2][TN][LDSW];   // transposed: Bs[buf][n][k]

    const int tid    = threadIdx.x;
    const int lane   = tid & 31;
    const int waveId = tid >> 5;      // 0..7
    const int wRow   = waveId >> 1;   // 0..3
    const int wCol   = waveId & 1;    // 0..1
    const int m0     = blockIdx.x * TM;
    const int n0     = blockIdx.y * TN;
    const int laneHi = lane >> 4;
    const int lane15 = lane & 15;

    const int ra = tid >> 2, qa = tid & 3;   // A: 4 threads/row, 8 floats each
    const int nb = tid >> 3, qb = tid & 7;   // B: 8 threads/row, 4 floats each
    const int mA = m0 + ra;

    const int* rowIdx = nullptr;
    const float* abase;
    if (GATHER) {
        const int vtx = mA & ((1 << LGV) - 1);
        rowIdx = idx + vtx * SP;
        abase  = x + ((mA >> LGV) << LGV) * CIN + qa * 8;   // frame base + lane offset
    } else {
        abase = x + mA * K + qa * 8;
    }
    const int nB = n0 + nb;
    const float* wrow = wt + nB * K + qb * 4;
    bool nv = true; if (NG) nv = (nB < Nout);

    v8f acc = {0.f, 0.f, 0.f, 0.f, 0.f, 0.f, 0.f, 0.f};

    auto stage = [&](int chunk, int buf) {
        float4 a0, a1;
        if (GATHER) {
            constexpr int CPS = CIN / KC;              // chunks per spiral entry
            const int s  = chunk / CPS;                // folds to shifts
            const int c0 = (chunk % CPS) * KC;
            const float* p = abase + rowIdx[s] * CIN + c0;
            a0 = *(const float4*)p; a1 = *(const float4*)(p + 4);
        } else {
            const float* p = abase + chunk * KC;
            a0 = *(const float4*)p; a1 = *(const float4*)(p + 4);
        }
        unsigned* da = (unsigned*)&As[buf][ra][qa * 8];
        da[0] = pkh(a0.x, a0.y); da[1] = pkh(a0.z, a0.w);
        da[2] = pkh(a1.x, a1.y); da[3] = pkh(a1.z, a1.w);
        float4 bv = {0.f, 0.f, 0.f, 0.f};
        if (nv) bv = *(const float4*)(wrow + chunk * KC);
        unsigned* db = (unsigned*)&Bs[buf][nb][qb * 4];
        db[0] = pkh(bv.x, bv.y); db[1] = pkh(bv.z, bv.w);
    };

    constexpr int NITER = K / KC;
    stage(0, 0);
    __syncthreads();
    #pragma unroll 2
    for (int it = 0; it < NITER; ++it) {
        if (it + 1 < NITER) stage(it + 1, (it + 1) & 1);
        const int cur = it & 1;
        FragH a, b;
        #pragma unroll
        for (int j = 0; j < 8; ++j) {
            // A 16x32 f16: VGPR j holds K = 2j (+8 if j>=4), +8 for lanes 16-31
            int ka = 2 * j + ((j >= 4) ? 8 : 0) + 8 * laneHi;
            a.u[j] = *(const unsigned*)&As[cur][wRow * 16 + lane15][ka];
            // B 32x16 f16: VGPR j holds K = 2j (+8 if j>=4), +8 for lanes 16-31
            int kb = 2 * j + ((j >= 4) ? 8 : 0) + 8 * laneHi;
            b.u[j] = *(const unsigned*)&Bs[cur][wCol * 16 + lane15][kb];
        }
        acc = __builtin_amdgcn_wmma_f32_16x16x32_f16(false, a.v, false, b.v,
                                                     (short)0, acc, false, false);
        __syncthreads();
    }

    #pragma unroll
    for (int r = 0; r < 8; ++r) {
        const int m = m0 + wRow * 16 + r + 8 * laneHi;
        const int n = n0 + wCol * 16 + lane15;
        if (!NG || n < Nout) {
            float v = acc[r] + bias[n];
            if (ELU) v = (v > 0.f) ? v : expm1f(v);
            if (RES) v += res[(m & ((1 << LGV) - 1)) * Nout + n];
            out[m * Nout + n] = v;
        }
    }
}

// Branchless small WMMA GEMM for the two edge layers:
//   enc0   : GATHER (batch 1, vtx == m), K=27, CIN=3, M=8192, N=32
//   enc_lat: dense, M=1, K=4096, N=64
// Out-of-range handled by clamped addresses + value selects (no exec-mask branches).
template<bool GATHER, int CIN>
__global__ __launch_bounds__(128)
void wmma_small(const float* __restrict__ x, const int* __restrict__ idx,
                const float* __restrict__ wt, const float* __restrict__ bias,
                float* __restrict__ out, int M, int Nout, int K, int elu)
{
    __shared__ _Float16 As[32][LDSW];
    __shared__ _Float16 Bs[32][LDSW];
    const int tid = threadIdx.x;
    const int lane = tid & 31;
    const int waveId = tid >> 5;
    const int wRow = waveId >> 1, wCol = waveId & 1;
    const int m0 = blockIdx.x * 32, n0 = blockIdx.y * 32;
    const int laneHi = lane >> 4, lane15 = lane & 15;

    v8f acc = {0.f, 0.f, 0.f, 0.f, 0.f, 0.f, 0.f, 0.f};
    for (int k0 = 0; k0 < K; k0 += KC) {
        #pragma unroll
        for (int e = 0; e < 4; ++e) {                  // A: 512 dword slots
            int linear = e * 128 + tid;
            int r  = linear >> 4;
            int kp = (linear & 15) * 2;
            int m  = m0 + r;
            int mm = (m < M) ? m : (M - 1);
            int k  = k0 + kp;
            int kc0 = (k     < K) ? k     : (K - 1);
            int kc1 = (k + 1 < K) ? k + 1 : (K - 1);
            float v0, v1;
            if (GATHER) {
                int s0 = kc0 / CIN, c0 = kc0 - s0 * CIN;
                int s1 = kc1 / CIN, c1 = kc1 - s1 * CIN;
                v0 = x[idx[mm * SP + s0] * CIN + c0];
                v1 = x[idx[mm * SP + s1] * CIN + c1];
            } else {
                v0 = x[mm * K + kc0];
                v1 = x[mm * K + kc1];
            }
            v0 = (m < M && k     < K) ? v0 : 0.f;
            v1 = (m < M && k + 1 < K) ? v1 : 0.f;
            *(unsigned*)&As[r][kp] = pkh(v0, v1);
        }
        #pragma unroll
        for (int e = 0; e < 4; ++e) {                  // B: 512 dword slots
            int linear = e * 128 + tid;
            int n  = linear >> 4;
            int kp = (linear & 15) * 2;
            int nn = n0 + n;
            int nc = (nn < Nout) ? nn : (Nout - 1);
            int k  = k0 + kp;
            int kc0 = (k     < K) ? k     : (K - 1);
            int kc1 = (k + 1 < K) ? k + 1 : (K - 1);
            float v0 = wt[nc * K + kc0];
            float v1 = wt[nc * K + kc1];
            v0 = (nn < Nout && k     < K) ? v0 : 0.f;
            v1 = (nn < Nout && k + 1 < K) ? v1 : 0.f;
            *(unsigned*)&Bs[n][kp] = pkh(v0, v1);
        }
        __syncthreads();
        FragH a, b;
        #pragma unroll
        for (int j = 0; j < 8; ++j) {
            int ka = 2 * j + ((j >= 4) ? 8 : 0) + 8 * laneHi;
            a.u[j] = *(const unsigned*)&As[wRow * 16 + lane15][ka];
            int kb = 2 * j + ((j >= 4) ? 8 : 0) + 8 * laneHi;
            b.u[j] = *(const unsigned*)&Bs[wCol * 16 + lane15][kb];
        }
        acc = __builtin_amdgcn_wmma_f32_16x16x32_f16(false, a.v, false, b.v,
                                                     (short)0, acc, false, false);
        __syncthreads();
    }
    #pragma unroll
    for (int r = 0; r < 8; ++r) {
        int m = m0 + wRow * 16 + r + 8 * laneHi;
        int n = n0 + wCol * 16 + lane15;
        if (m < M && n < Nout) {
            float v = acc[r] + bias[n];
            if (elu) v = (v > 0.f) ? v : expm1f(v);
            out[m * Nout + n] = v;
        }
    }
}

// wt[n*K + k] = w[k*N + n]
__global__ __launch_bounds__(256)
void transpose_kernel(const float* __restrict__ w, float* __restrict__ wt, int K, int N)
{
    int i = blockIdx.x * 256 + threadIdx.x;
    if (i >= K * N) return;
    int k = i / N, n = i - k * N;
    wt[n * K + k] = w[i];
}

// 3-nnz COO pool, float4 over channels; C and Vout are powers of two.
// out[b,r,ch] = sum_{j<3} val[3r+j] * x[b, col[3r+j], ch]
__global__ __launch_bounds__(256)
void pool3_kernel(const float* __restrict__ x, const int* __restrict__ col,
                  const float* __restrict__ val, float* __restrict__ out,
                  int total4, int lgC4, int lgVout, int Vin)
{
    int i = blockIdx.x * 256 + threadIdx.x;
    if (i >= total4) return;
    const int C4   = 1 << lgC4;
    const int ch   = i & (C4 - 1);
    const int r    = (i >> lgC4) & ((1 << lgVout) - 1);
    const int b    = i >> (lgC4 + lgVout);
    const float4* xb = (const float4*)x + b * Vin * C4;
    float4 acc = {0.f, 0.f, 0.f, 0.f};
    #pragma unroll
    for (int j = 0; j < 3; ++j) {
        const int   cc = col[3 * r + j];
        const float vv = val[3 * r + j];
        const float4 xv = xb[cc * C4 + ch];
        acc.x += vv * xv.x; acc.y += vv * xv.y;
        acc.z += vv * xv.z; acc.w += vv * xv.w;
    }
    ((float4*)out)[i] = acc;
}

__global__ __launch_bounds__(128)
void concat_latent_kernel(const float* __restrict__ audioE,
                          const float* __restrict__ actorE,
                          float* __restrict__ latent, int T)
{
    int i = blockIdx.x * blockDim.x + threadIdx.x;
    if (i >= T * 128) return;
    int t = i >> 7, c = i & 127;
    latent[i] = (c < 64) ? audioE[t * 64 + c] : actorE[c - 64];
}

extern "C" void kernel_launch(void* const* d_in, const int* in_sizes, int n_in,
                              void* d_out, int out_size, void* d_ws, size_t ws_size,
                              hipStream_t stream)
{
    const int T = 64, LAT = 64;
    const int V0 = 8192, V1 = 2048, V2 = 512, V3 = 128, V4 = 32;
    const int OC0 = 32;

    const float* hidden  = (const float*)d_in[0];
    const float* actor   = (const float*)d_in[1];
    const float* audio_w = (const float*)d_in[2];
    const float* audio_b = (const float*)d_in[3];
    const float* enc_w[4] = {(const float*)d_in[4], (const float*)d_in[6],
                             (const float*)d_in[8], (const float*)d_in[10]};
    const float* enc_b[4] = {(const float*)d_in[5], (const float*)d_in[7],
                             (const float*)d_in[9], (const float*)d_in[11]};
    const float* enc_lat_w = (const float*)d_in[12];
    const float* enc_lat_b = (const float*)d_in[13];
    const float* dec_lin_w = (const float*)d_in[14];
    const float* dec_lin_b = (const float*)d_in[15];
    const float* dec_w[4] = {(const float*)d_in[16], (const float*)d_in[18],
                             (const float*)d_in[20], (const float*)d_in[22]};
    const float* dec_b[4] = {(const float*)d_in[17], (const float*)d_in[19],
                             (const float*)d_in[21], (const float*)d_in[23]};
    const float* dec_out_w = (const float*)d_in[24];
    const float* dec_out_b = (const float*)d_in[25];
    const int* spiral[4] = {(const int*)d_in[26], (const int*)d_in[27],
                            (const int*)d_in[28], (const int*)d_in[29]};
    const int*   dn_col[4]; const float* dn_val[4];
    const int*   up_col[4]; const float* up_val[4];
    for (int i = 0; i < 4; ++i) {
        int base = 30 + 6 * i;
        dn_col[i] = (const int*)  d_in[base + 1];
        dn_val[i] = (const float*)d_in[base + 2];
        up_col[i] = (const int*)  d_in[base + 4];
        up_val[i] = (const float*)d_in[base + 5];
    }

    // ---- workspace carve ----
    char* ws = (char*)d_ws;
    float* slotB  = (float*)ws;                                   // <=128 MB
    float* slotA  = (float*)(ws + (size_t)134217728);             // <=64 MB
    float* encA   = (float*)(ws + (size_t)134217728 + 67108864);
    float* encB   = encA + V0 * OC0;
    float* audioE = encB + V0 * OC0;
    float* actorE = audioE + T * LAT;
    float* latent = actorE + LAT;
    float* p      = latent + T * 2 * LAT;       // transposed-weight arena (~5 MB)
    float* wt_enc0 = p; p += 27 * 32;
    float* wt_enc1 = p; p += 288 * 64;
    float* wt_enc2 = p; p += 576 * 64;
    float* wt_enc3 = p; p += 576 * 128;
    float* wt_lat  = p; p += 4096 * 64;
    float* wt_aud  = p; p += 768 * 64;
    float* wt_dlin = p; p += 128 * 4096;
    float* wt_dec0 = p; p += 1152 * 128;
    float* wt_dec1 = p; p += 1152 * 64;
    float* wt_dec2 = p; p += 576 * 64;
    float* wt_dec3 = p; p += 576 * 32;
    float* wt_dout = p; p += 288 * 3;

    auto tr = [&](const float* w, float* wt, int K, int N) {
        int tot = K * N;
        transpose_kernel<<<dim3((tot + 255) / 256), dim3(256), 0, stream>>>(w, wt, K, N);
    };
    tr(enc_w[0], wt_enc0, 27, 32);     tr(enc_w[1], wt_enc1, 288, 64);
    tr(enc_w[2], wt_enc2, 576, 64);    tr(enc_w[3], wt_enc3, 576, 128);
    tr(enc_lat_w, wt_lat, 4096, 64);   tr(audio_w, wt_aud, 768, 64);
    tr(dec_lin_w, wt_dlin, 128, 4096); tr(dec_w[0], wt_dec0, 1152, 128);
    tr(dec_w[1], wt_dec1, 1152, 64);   tr(dec_w[2], wt_dec2, 576, 64);
    tr(dec_w[3], wt_dec3, 576, 32);    tr(dec_out_w, wt_dout, 288, 3);

    auto pool = [&](const float* x, const int* col, const float* val, float* o,
                    int B, int lgVout, int Vin, int lgC4) {
        int total4 = B << (lgVout + lgC4);
        pool3_kernel<<<dim3((total4 + 255) / 256), dim3(256), 0, stream>>>(
            x, col, val, o, total4, lgC4, lgVout, Vin);
    };
    const dim3 blk(256);

    // ================= encoder (batch 1) =================
    wmma_small<true, 3><<<dim3(V0 / 32, 1), dim3(128), 0, stream>>>(
        actor, spiral[0], wt_enc0, enc_b[0], encA, V0, OC0, 27, 1);
    pool(encA, dn_col[0], dn_val[0], encB, 1, 11, V0, 3);       // 2048 x 32ch
    wmma_conv<288, 32, 11, true, true, false, false>
        <<<dim3(V1 / TM, 2), blk, 0, stream>>>(encB, spiral[1], wt_enc1, enc_b[1], nullptr, encA, 64);
    pool(encA, dn_col[1], dn_val[1], encB, 1, 9, V1, 4);        // 512 x 64ch
    wmma_conv<576, 64, 9, true, true, false, false>
        <<<dim3(V2 / TM, 2), blk, 0, stream>>>(encB, spiral[2], wt_enc2, enc_b[2], nullptr, encA, 64);
    pool(encA, dn_col[2], dn_val[2], encB, 1, 7, V2, 4);        // 128 x 64ch
    wmma_conv<576, 64, 7, true, true, false, false>
        <<<dim3(V3 / TM, 4), blk, 0, stream>>>(encB, spiral[3], wt_enc3, enc_b[3], nullptr, encA, 128);
    pool(encA, dn_col[3], dn_val[3], encB, 1, 5, V3, 5);        // 32 x 128ch
    wmma_small<false, 1><<<dim3(1, 2), dim3(128), 0, stream>>>(
        encB, nullptr, wt_lat, enc_lat_b, actorE, 1, LAT, 4096, 0);

    // ================= audio + latent =================
    wmma_conv<768, 32, 0, false, false, false, false>
        <<<dim3(1, 2), blk, 0, stream>>>(hidden, nullptr, wt_aud, audio_b, nullptr, audioE, 64);
    concat_latent_kernel<<<dim3((T * 128 + 127) / 128), dim3(128), 0, stream>>>(
        audioE, actorE, latent, T);
    wmma_conv<128, 32, 0, false, false, false, false>
        <<<dim3(1, 128), blk, 0, stream>>>(latent, nullptr, wt_dlin, dec_lin_b, nullptr, slotA, 4096);

    // ================= decoder (batched over T frames) =================
    pool(slotA, up_col[3], up_val[3], slotB, T, 7, V4, 5);      // -> [T,128,128]
    wmma_conv<1152, 128, 7, true, true, false, false>
        <<<dim3(T * V3 / TM, 4), blk, 0, stream>>>(slotB, spiral[3], wt_dec0, dec_b[0], nullptr, slotA, 128);
    pool(slotA, up_col[2], up_val[2], slotB, T, 9, V3, 5);      // -> [T,512,128]
    wmma_conv<1152, 128, 9, true, true, false, false>
        <<<dim3(T * V2 / TM, 2), blk, 0, stream>>>(slotB, spiral[2], wt_dec1, dec_b[1], nullptr, slotA, 64);
    pool(slotA, up_col[1], up_val[1], slotB, T, 11, V2, 4);     // -> [T,2048,64]
    wmma_conv<576, 64, 11, true, true, false, false>
        <<<dim3(T * V1 / TM, 2), blk, 0, stream>>>(slotB, spiral[1], wt_dec2, dec_b[2], nullptr, slotA, 64);
    pool(slotA, up_col[0], up_val[0], slotB, T, 13, V1, 4);     // -> [T,8192,64]
    wmma_conv<576, 64, 13, true, true, false, false>
        <<<dim3(T * V0 / TM, 1), blk, 0, stream>>>(slotB, spiral[0], wt_dec3, dec_b[3], nullptr, slotA, 32);
    wmma_conv<288, 32, 13, true, false, true, true>
        <<<dim3(T * V0 / TM, 1), blk, 0, stream>>>(slotA, spiral[0], wt_dout, dec_out_b, actor,
                                                   (float*)d_out, 3);
}
